// AttentionSequencePoolingLayer_1391569404549
// MI455X (gfx1250) — matrix operand order repaired and checked by reading, not compiled, not run on hardware
//
#include <hip/hip_runtime.h>
#include <math.h>

// ---------------- problem constants ----------------
constexpr int cB  = 2048;
constexpr int cT  = 200;
constexpr int cE  = 64;
constexpr int cH1 = 64;
constexpr int cH2 = 16;
constexpr int cDIN = 4 * cE;          // 256
constexpr int cBT = cB * cT;          // 409600 (multiple of 16)
constexpr int cNT = cBT / 16;         // 25600 row-tiles
constexpr int cWPB = 4;               // waves (tiles) per block
constexpr int cNBLK = cNT / cWPB;     // 6400 blocks for GEMM stages

constexpr int cPAD1 = cDIN + 8;       // 264: LDS row stride (bf16) for K=256 tiles
constexpr int cPADW2 = cH1 + 8;       // 72:  LDS row stride for K=64 tiles
constexpr int cKP = 224;              // K padded for T=200 einsum
constexpr int cPADT = 240;            // LDS row stride for K<=224 tiles

typedef __attribute__((ext_vector_type(16))) __bf16 v16bf;
typedef __attribute__((ext_vector_type(8)))  __bf16 v8bf;
typedef __attribute__((ext_vector_type(8)))  float  v8f;

union Frag16 { v16bf v; v8bf h[2]; };

__device__ __forceinline__ unsigned pk2(float lo, float hi) {
    union { __bf16 b; unsigned short u; } x, y;
    x.b = (__bf16)lo; y.b = (__bf16)hi;
    return ((unsigned)y.u << 16) | (unsigned)x.u;
}

// Load one A/B fragment whose K-runs are contiguous in LDS.
// A fragment: 16 consecutive K at base (matches B fragment order)
__device__ __forceinline__ v16bf load_frag_a(const __bf16* p) {
    Frag16 f;
    f.h[0] = *(const v8bf*)(p);
    f.h[1] = *(const v8bf*)(p + 8);
    return f.v;
}
// B fragment: 16 consecutive K at base (contiguous k = 16*half + s, same order as A)
__device__ __forceinline__ v16bf load_frag_b(const __bf16* p) {
    Frag16 f;
    f.h[0] = *(const v8bf*)(p);
    f.h[1] = *(const v8bf*)(p + 8);
    return f.v;
}

// ---------------- stage 1: att_in @ W1 + b1, with Dice-stat partials ----------------
__global__ __launch_bounds__(128)
void k1_gemm1(const float* __restrict__ q, const float* __restrict__ hist,
              const float* __restrict__ W1, const float* __restrict__ b1,
              float* __restrict__ X1, float* __restrict__ p1S, float* __restrict__ p1Q) {
    __shared__ alignas(16) __bf16 sW1p[cH1][cPAD1];      // [n][k]  ~33 KB
    __shared__ alignas(16) __bf16 sA[cWPB][16][cPAD1];   // [row][feature] ~33.8 KB
    __shared__ float wS[cWPB][cH1];
    __shared__ float wQs[cWPB][cH1];

    const int tid = threadIdx.x;
    const int w = tid >> 5, lane = tid & 31;

    // W1 -> transposed bf16 LDS (coalesced global reads; one-time LDS scatter)
    for (int i = tid; i < cDIN * cH1; i += 128) {
        int k = i >> 6, n = i & 63;
        sW1p[n][k] = (__bf16)W1[i];
    }

    // build att_in tile (q, h, q-h, q*h) as packed bf16, K-contiguous per row
    const int tile = blockIdx.x * cWPB + w;
    const int R = tile * 16;
    for (int m = 0; m < 16; ++m) {
        int r = R + m;
        int b = r / cT, t = r - b * cT;
        float2 qv = ((const float2*)(q + (size_t)b * cE))[lane];
        float2 hv = ((const float2*)(hist + (size_t)(b * cT + t) * cE))[lane];
        unsigned* rowp = (unsigned*)&sA[w][m][0];
        rowp[0 * 32 + lane] = pk2(qv.x, qv.y);
        rowp[1 * 32 + lane] = pk2(hv.x, hv.y);
        rowp[2 * 32 + lane] = pk2(qv.x - hv.x, qv.y - hv.y);
        rowp[3 * 32 + lane] = pk2(qv.x * hv.x, qv.y * hv.y);
    }
    __syncthreads();

    const int half = lane >> 4, mrow = lane & 15, nloc = lane & 15;
    v8f acc[4] = {};

#pragma unroll
    for (int kk = 0; kk < 8; ++kk) {           // K = 256 in steps of 32
        v16bf a = load_frag_a(&sA[w][mrow][kk * 32 + half * 16]);
#pragma unroll
        for (int nt = 0; nt < 4; ++nt) {
            v16bf bb = load_frag_b(&sW1p[nt * 16 + nloc][kk * 32 + half * 16]);
            acc[nt] = __builtin_amdgcn_wmma_f32_16x16x32_bf16(
                false, a, false, bb, (short)0, acc[nt], false, false);
        }
    }

#pragma unroll
    for (int nt = 0; nt < 4; ++nt) {
        int n = nt * 16 + nloc;
        float bias = b1[n];
        float s = 0.f, sq = 0.f;
#pragma unroll
        for (int v = 0; v < 8; ++v) {
            float x = acc[nt][v] + bias;
            int m = v + half * 8;               // C layout: M = v (+8 for hi half)
            X1[(size_t)(R + m) * cH1 + n] = x;
            s += x; sq += x * x;
        }
        s  += __shfl_xor(s, 16, 32);
        sq += __shfl_xor(sq, 16, 32);
        if (half == 0) { wS[w][n] = s; wQs[w][n] = sq; }
    }
    __syncthreads();
    if (tid < cH1) {    // fixed-order combine -> deterministic
        float s  = wS[0][tid] + wS[1][tid] + wS[2][tid] + wS[3][tid];
        float sq = wQs[0][tid] + wQs[1][tid] + wQs[2][tid] + wQs[3][tid];
        p1S[blockIdx.x * cH1 + tid] = s;
        p1Q[blockIdx.x * cH1 + tid] = sq;
    }
}

// ---------------- stage 2: deterministic stats reduce (layer 1) ----------------
__global__ void k2_reduce1(const float* __restrict__ p1S, const float* __restrict__ p1Q,
                           float* __restrict__ m1, float* __restrict__ r1) {
    int f = threadIdx.x;
    if (f >= cH1) return;
    float s = 0.f, qq = 0.f;
    for (int blk = 0; blk < cNBLK; ++blk) { s += p1S[blk * cH1 + f]; qq += p1Q[blk * cH1 + f]; }
    float mean = s / (float)cBT;
    float var = qq / (float)cBT - mean * mean;
    m1[f] = mean;
    r1[f] = rsqrtf(var + 1e-8f);
}

// ---------------- stage 3: dice(X1) @ W2 + b2, with stat partials ----------------
__global__ __launch_bounds__(128)
void k3_gemm2(const float* __restrict__ X1, const float* __restrict__ W2,
              const float* __restrict__ b2, const float* __restrict__ m1,
              const float* __restrict__ r1, const float* __restrict__ al1,
              float* __restrict__ X2, float* __restrict__ p2S, float* __restrict__ p2Q) {
    __shared__ alignas(16) __bf16 sW2p[cH2][cPADW2];      // [n][k]
    __shared__ alignas(16) __bf16 sHd[cWPB][16][cPADW2];  // dice(X1) rows, K-contiguous
    __shared__ float sm[cH1], sr[cH1], sa[cH1];
    __shared__ float wS[cWPB][cH2], wQs[cWPB][cH2];

    const int tid = threadIdx.x;
    const int w = tid >> 5, lane = tid & 31;

    for (int i = tid; i < cH1 * cH2; i += 128) {
        int k = i >> 4, n = i & 15;
        sW2p[n][k] = (__bf16)W2[i];
    }
    if (tid < cH1) { sm[tid] = m1[tid]; sr[tid] = r1[tid]; sa[tid] = al1[tid]; }
    __syncthreads();   // sm/sr/sa used below by all threads

    const int tile = blockIdx.x * cWPB + w;
    const int R = tile * 16;
    for (int m = 0; m < 16; ++m) {
        float2 xv = ((const float2*)(X1 + (size_t)(R + m) * cH1))[lane];
        int e0 = lane * 2;
        float xn0 = (xv.x - sm[e0]) * sr[e0];
        float p0 = 1.f / (1.f + __expf(-xn0));
        float h0 = xv.x * (p0 + (1.f - p0) * sa[e0]);
        float xn1 = (xv.y - sm[e0 + 1]) * sr[e0 + 1];
        float p1 = 1.f / (1.f + __expf(-xn1));
        float h1 = xv.y * (p1 + (1.f - p1) * sa[e0 + 1]);
        ((unsigned*)&sHd[w][m][0])[lane] = pk2(h0, h1);
    }
    __syncthreads();

    const int half = lane >> 4, mrow = lane & 15, nloc = lane & 15;
    v8f acc = {};
#pragma unroll
    for (int kk = 0; kk < 2; ++kk) {           // K = 64
        v16bf a  = load_frag_a(&sHd[w][mrow][kk * 32 + half * 16]);
        v16bf bb = load_frag_b(&sW2p[nloc][kk * 32 + half * 16]);
        acc = __builtin_amdgcn_wmma_f32_16x16x32_bf16(
            false, a, false, bb, (short)0, acc, false, false);
    }

    float bias = b2[nloc];
    float s = 0.f, sq = 0.f;
#pragma unroll
    for (int v = 0; v < 8; ++v) {
        float x = acc[v] + bias;
        int m = v + half * 8;
        X2[(size_t)(R + m) * cH2 + nloc] = x;
        s += x; sq += x * x;
    }
    s  += __shfl_xor(s, 16, 32);
    sq += __shfl_xor(sq, 16, 32);
    if (half == 0) { wS[w][nloc] = s; wQs[w][nloc] = sq; }
    __syncthreads();
    if (tid < cH2) {
        float ss  = wS[0][tid] + wS[1][tid] + wS[2][tid] + wS[3][tid];
        float sqq = wQs[0][tid] + wQs[1][tid] + wQs[2][tid] + wQs[3][tid];
        p2S[blockIdx.x * cH2 + tid] = ss;
        p2Q[blockIdx.x * cH2 + tid] = sqq;
    }
}

// ---------------- stage 4: stats reduce (layer 2) ----------------
__global__ void k4_reduce2(const float* __restrict__ p2S, const float* __restrict__ p2Q,
                           float* __restrict__ m2, float* __restrict__ r2) {
    int f = threadIdx.x;
    if (f >= cH2) return;
    float s = 0.f, qq = 0.f;
    for (int blk = 0; blk < cNBLK; ++blk) { s += p2S[blk * cH2 + f]; qq += p2Q[blk * cH2 + f]; }
    float mean = s / (float)cBT;
    float var = qq / (float)cBT - mean * mean;
    m2[f] = mean;
    r2[f] = rsqrtf(var + 1e-8f);
}

// ---------------- stage 5: dice2 + mask + softmax + score@hist (per batch) ----------------
__global__ __launch_bounds__(128)
void k5_softmax_out(const float* __restrict__ hist, const int* __restrict__ lens,
                    const float* __restrict__ X2, const float* __restrict__ m2,
                    const float* __restrict__ r2, const float* __restrict__ al2,
                    float* __restrict__ out) {
    __shared__ alignas(16) __bf16 sHistT[cE][cPADT];   // [e][t], t zero-padded to 240
    __shared__ alignas(16) __bf16 sScB[cH2][cPADT];    // [h][t] bf16 scores
    __shared__ float sSc[cT * cH2];                    // masked logits f32 ([t][h])
    __shared__ float red[8][cH2];
    __shared__ float colmax[cH2], colsum[cH2];

    const int tid = threadIdx.x, b = blockIdx.x;

    // zero the padded bf16 tiles (dword stores), then sync before 16-bit fills
    {
        unsigned* z0 = (unsigned*)&sHistT[0][0];
        for (int i = tid; i < cE * cPADT / 2; i += 128) z0[i] = 0u;
        unsigned* z1 = (unsigned*)&sScB[0][0];
        for (int i = tid; i < cH2 * cPADT / 2; i += 128) z1[i] = 0u;
    }
    __syncthreads();

    // hist -> transposed bf16 [e][t]
    const float* hb = hist + (size_t)b * cT * cE;
    for (int i = tid; i < cT * cE; i += 128) {
        int t = i >> 6, e = i & 63;
        sHistT[e][t] = (__bf16)hb[i];
    }

    // dice2 + (reversed) mask -> f32 masked logits
    const int len = lens[b];
    for (int i = tid; i < cT * cH2; i += 128) {
        int t = i >> 4, h = i & 15;
        float x = X2[((size_t)b * cT + t) * cH2 + h];
        float xn = (x - m2[h]) * r2[h];
        float p = 1.f / (1.f + __expf(-xn));
        float v = x * (p + (1.f - p) * al2[h]);
        // reference: where(t < len, 1e-9, logits) — valid slots get the constant
        sSc[i] = (t < len) ? 1e-9f : v;
    }
    __syncthreads();

    // deterministic segmented softmax over T per column h
    const int h = tid & 15, seg = tid >> 4;            // 8 segments x 25 steps
    const int t0 = seg * 25, t1 = t0 + 25;
    float mx = -3.402823466e38f;
    for (int t = t0; t < t1; ++t) mx = fmaxf(mx, sSc[t * cH2 + h]);
    red[seg][h] = mx;
    __syncthreads();
    if (seg == 0) {
        float m = red[0][h];
        for (int j = 1; j < 8; ++j) m = fmaxf(m, red[j][h]);
        colmax[h] = m;
    }
    __syncthreads();
    const float cm = colmax[h];
    float smv = 0.f;
    for (int t = t0; t < t1; ++t) smv += __expf(sSc[t * cH2 + h] - cm);
    red[seg][h] = smv;
    __syncthreads();
    if (seg == 0) {
        float s2 = 0.f;
        for (int j = 0; j < 8; ++j) s2 += red[j][h];
        colsum[h] = s2;
    }
    __syncthreads();
    const float inv = 1.f / colsum[h];
    for (int t = t0; t < t1; ++t)
        sScB[h][t] = (__bf16)(__expf(sSc[t * cH2 + h] - cm) * inv);
    __syncthreads();

    // einsum: out[h][e] = sum_t score[h][t]*hist[t][e]; M=16, N=64 (4 waves), K=200->224
    const int w = tid >> 5, lane = tid & 31;
    const int half = lane >> 4, mrow = lane & 15, nloc = lane & 15;
    v8f acc = {};
#pragma unroll
    for (int kk = 0; kk < cKP / 32; ++kk) {
        v16bf a  = load_frag_a(&sScB[mrow][kk * 32 + half * 16]);
        v16bf bb = load_frag_b(&sHistT[w * 16 + nloc][kk * 32 + half * 16]);
        acc = __builtin_amdgcn_wmma_f32_16x16x32_bf16(
            false, a, false, bb, (short)0, acc, false, false);
    }
#pragma unroll
    for (int v = 0; v < 8; ++v) {
        int m = v + half * 8;
        out[(size_t)b * cH2 * cE + (size_t)m * cE + w * 16 + nloc] = acc[v];
    }
}

// ---------------- launcher ----------------
extern "C" void kernel_launch(void* const* d_in, const int* in_sizes, int n_in,
                              void* d_out, int out_size, void* d_ws, size_t ws_size,
                              hipStream_t stream) {
    (void)in_sizes; (void)n_in; (void)out_size; (void)ws_size;
    const float* q    = (const float*)d_in[0];   // [B,1,E]
    const float* hist = (const float*)d_in[1];   // [B,T,E]
    const int*   lens = (const int*)d_in[2];     // [B] int32
    const float* W1   = (const float*)d_in[3];   // [256,64]
    const float* b1   = (const float*)d_in[4];
    const float* al1  = (const float*)d_in[5];
    const float* W2   = (const float*)d_in[6];   // [64,16]
    const float* b2   = (const float*)d_in[7];
    const float* al2  = (const float*)d_in[8];
    float* out = (float*)d_out;                  // [B,16,64]

    // workspace layout (floats); total ~33.8M floats (~135 MB)
    float* ws  = (float*)d_ws;
    float* X1  = ws;                                   // BT*H1
    float* X2  = X1 + (size_t)cBT * cH1;               // BT*H2
    float* p1S = X2 + (size_t)cBT * cH2;               // NBLK*H1
    float* p1Q = p1S + (size_t)cNBLK * cH1;
    float* p2S = p1Q + (size_t)cNBLK * cH1;            // NBLK*H2
    float* p2Q = p2S + (size_t)cNBLK * cH2;
    float* m1  = p2Q + (size_t)cNBLK * cH2;
    float* r1  = m1 + cH1;
    float* m2  = r1 + cH1;
    float* r2  = m2 + cH2;

    k1_gemm1<<<cNBLK, 128, 0, stream>>>(q, hist, W1, b1, X1, p1S, p1Q);
    k2_reduce1<<<1, 64, 0, stream>>>(p1S, p1Q, m1, r1);
    k3_gemm2<<<cNBLK, 128, 0, stream>>>(X1, W2, b2, m1, r1, al1, X2, p2S, p2Q);
    k4_reduce2<<<1, 32, 0, stream>>>(p2S, p2Q, m2, r2);
    k5_softmax_out<<<cB, 128, 0, stream>>>(hist, lens, X2, m2, r2, al2, out);
}
